// ConvModule_11690900980570
// MI455X (gfx1250) — matrix operand order repaired and checked by reading, not compiled, not run on hardware
//
#include <hip/hip_runtime.h>
#include <hip/hip_bf16.h>

// ---------------------------------------------------------------------------
// Problem constants
// ---------------------------------------------------------------------------
#define BB   16
#define TT   1024
#define CC   512
#define BT   (BB * TT)          // 16384 rows
#define C2   (2 * CC)           // 1024
#define KW   5

// LDS A-tile geometry: 64 rows x 32 K bf16, padded row stride 40 (80 bytes)
#define TILE_ROWS   64
#define TILE_STRIDE 40

typedef __attribute__((ext_vector_type(16))) __bf16        bf16x16;
typedef __attribute__((ext_vector_type(8)))  float         floatx8;
typedef __attribute__((ext_vector_type(4)))  unsigned int  uintx4;

union Frag {
    uintx4  u[2];
    bf16x16 v;
};

__device__ __forceinline__ unsigned short f2bf(float f) {
    unsigned int u = __float_as_uint(f);
    unsigned int r = (u + 0x7FFFu + ((u >> 16) & 1u)) >> 16;   // RNE
    return (unsigned short)r;
}

__device__ __forceinline__ floatx8 fzero8() {
    floatx8 z = {0.f, 0.f, 0.f, 0.f, 0.f, 0.f, 0.f, 0.f};
    return z;
}

// Async global->LDS copy of 16 bytes per lane (CDNA5 ASYNCcnt path).
// Takes the real LDS destination pointer so the shared array is captured
// (ptrtoint) and the asm "memory" clobber is understood to write it.
// Low 32 bits of a shared-aperture generic address == LDS byte address.
__device__ __forceinline__ void async_copy16(void* lds_ptr, const void* gptr) {
    unsigned off = (unsigned)(size_t)lds_ptr;
    asm volatile("global_load_async_to_lds_b128 %0, %1, off"
                 :: "v"(off), "v"(gptr)
                 : "memory");
}
__device__ __forceinline__ void wait_async0() {
    asm volatile("s_wait_asynccnt 0" ::: "memory");
}

// ---------------------------------------------------------------------------
// Kernel 0: weight conversion to bf16 (+ transpose of w2 to [K][Cout][Cin])
// ---------------------------------------------------------------------------
__global__ __launch_bounds__(256) void k_prep(
    const float* __restrict__ w1, const float* __restrict__ w2,
    const float* __restrict__ w3,
    unsigned short* __restrict__ w1b, unsigned short* __restrict__ w2t,
    unsigned short* __restrict__ w3b)
{
    size_t i = (size_t)blockIdx.x * 256 + threadIdx.x;
    if (i < (size_t)C2 * CC)  w1b[i] = f2bf(w1[i]);             // [1024,512]
    if (i < (size_t)CC * CC)  w3b[i] = f2bf(w3[i]);             // [512,512]
    if (i < (size_t)KW * C2 * C2) {                             // transpose
        int k   = (int)(i >> 20);
        int rem = (int)(i & 1048575u);
        int o   = rem >> 10;
        int c   = rem & 1023;
        w2t[i]  = f2bf(w2[((size_t)k << 20) + ((size_t)c << 10) + o]);
    }
}

// ---------------------------------------------------------------------------
// Kernel 1: LayerNorm -> bf16 (one wave per row)
// ---------------------------------------------------------------------------
__global__ __launch_bounds__(256) void k_ln(
    const float* __restrict__ x, const float* __restrict__ g,
    const float* __restrict__ b, unsigned short* __restrict__ xn)
{
    int row  = blockIdx.x * 8 + (threadIdx.x >> 5);
    int lane = threadIdx.x & 31;
    const float* xr = x + (size_t)row * CC;

    float s = 0.f, s2 = 0.f;
    #pragma unroll
    for (int i = 0; i < 16; ++i) {
        float v = xr[lane + 32 * i];
        s += v; s2 += v * v;
    }
    #pragma unroll
    for (int off = 16; off > 0; off >>= 1) {
        s  += __shfl_xor(s,  off, 32);
        s2 += __shfl_xor(s2, off, 32);
    }
    float mu   = s * (1.f / CC);
    float var  = s2 * (1.f / CC) - mu * mu;
    float rstd = rsqrtf(var + 1e-5f);

    unsigned short* orow = xn + (size_t)row * CC;
    #pragma unroll
    for (int i = 0; i < 16; ++i) {
        int c = lane + 32 * i;
        orow[c] = f2bf((xr[c] - mu) * rstd * g[c] + b[c]);
    }
}

// ---------------------------------------------------------------------------
// Kernel 2: GEMM1 [BT,512]x[512->1024] + bias + SiLU  -> h1 bf16
// 64x128 tile, 8 waves as 2x4, 2x2 WMMA per wave.
// Double-buffered LDS A-tiles filled by async global->LDS copies.
// ---------------------------------------------------------------------------
__global__ __launch_bounds__(256) void k_gemm1_silu(
    const unsigned short* __restrict__ xn,   // [BT,512] bf16
    const unsigned short* __restrict__ w1b,  // [1024,512] bf16
    const float* __restrict__ b1,            // [1024]
    unsigned short* __restrict__ h1)         // [BT,1024] bf16
{
    __shared__ __align__(16) unsigned short As[2][TILE_ROWS][TILE_STRIDE];

    const int tid  = threadIdx.x;
    const int wave = tid >> 5, lane = tid & 31;
    const int wm = wave >> 2, wn = wave & 3;
    const int lm = lane & 15, lh = lane >> 4;
    const int m0 = blockIdx.x * 64;
    const int n0 = blockIdx.y * 128;
    const int srow = tid >> 2, sq = tid & 3;

    const unsigned short* srcBase = xn + (size_t)(m0 + srow) * CC + sq * 8;

    floatx8 acc[2][2];
    #pragma unroll
    for (int i = 0; i < 2; ++i)
        #pragma unroll
        for (int j = 0; j < 2; ++j) acc[i][j] = fzero8();

    // prologue: chunk 0 -> buffer 0
    async_copy16(&As[0][srow][sq * 8], srcBase);
    wait_async0();
    __syncthreads();

    const int NCH = CC / 32;                 // 16 chunks
    for (int c = 0; c < NCH; ++c) {
        const int cur = c & 1;
        if (c + 1 < NCH)                     // overlap next stage with compute
            async_copy16(&As[cur ^ 1][srow][sq * 8], srcBase + (c + 1) * 32);

        Frag a[2], bfr[2];
        #pragma unroll
        for (int mi = 0; mi < 2; ++mi) {
            int r = wm * 32 + mi * 16 + lm;
            a[mi].u[0] = *(const uintx4*)(&As[cur][r][lh * 8]);
            a[mi].u[1] = *(const uintx4*)(&As[cur][r][16 + lh * 8]);
        }
        #pragma unroll
        for (int ni = 0; ni < 2; ++ni) {
            int o = n0 + wn * 32 + ni * 16 + lm;
            const unsigned short* wp = w1b + (size_t)o * CC + c * 32 + lh * 8;
            __builtin_prefetch(wp + 32, 0, 0);
            bfr[ni].u[0] = *(const uintx4*)(wp);
            bfr[ni].u[1] = *(const uintx4*)(wp + 16);
        }
        #pragma unroll
        for (int mi = 0; mi < 2; ++mi)
            #pragma unroll
            for (int ni = 0; ni < 2; ++ni)
                acc[mi][ni] = __builtin_amdgcn_wmma_f32_16x16x32_bf16(
                    false, a[mi].v, false, bfr[ni].v, (short)0, acc[mi][ni],
                    false, false);

        wait_async0();                       // my next-chunk copy landed
        __syncthreads();                     // publish buf, retire reads
    }

    #pragma unroll
    for (int mi = 0; mi < 2; ++mi)
        #pragma unroll
        for (int ni = 0; ni < 2; ++ni) {
            int col  = n0 + wn * 32 + ni * 16 + lm;
            float bi = b1[col];
            #pragma unroll
            for (int r = 0; r < 8; ++r) {
                int row = m0 + wm * 32 + mi * 16 + lh * 8 + r;
                float v = acc[mi][ni][r] + bi;
                float sv = v / (1.f + __expf(-v));           // SiLU
                h1[(size_t)row * C2 + col] = f2bf(sv);
            }
        }
}

// ---------------------------------------------------------------------------
// Kernel 3: conv1d as 5 shifted accumulated GEMMs + GLU + BatchNorm -> h2 bf16
// Flattened (k-tap, K-chunk) loop; async double-buffered A staging with
// zero-filled halo rows; each wave carries the o and o+512 accumulators.
// ---------------------------------------------------------------------------
__global__ __launch_bounds__(256) void k_conv_glu_bn(
    const unsigned short* __restrict__ h1,   // [BT,1024] bf16
    const unsigned short* __restrict__ w2t,  // [5][1024][1024] bf16 (o-major)
    const float* __restrict__ b2,
    const float* __restrict__ bn_g, const float* __restrict__ bn_b,
    const float* __restrict__ bn_mean, const float* __restrict__ bn_var,
    unsigned short* __restrict__ h2)         // [BT,512] bf16
{
    __shared__ __align__(16) unsigned short As[2][TILE_ROWS][TILE_STRIDE];

    const int tid  = threadIdx.x;
    const int wave = tid >> 5, lane = tid & 31;
    const int wm = wave >> 2, wn = wave & 3;
    const int lm = lane & 15, lh = lane >> 4;
    const int m0 = blockIdx.x * 64;
    const int bb = m0 >> 10;                 // batch
    const int t0 = m0 & (TT - 1);            // position inside batch
    const int n0 = blockIdx.y * 128;         // o in [n0,n0+128) and +512
    const int srow = tid >> 2, sq = tid & 3;

    floatx8 acc[2][2][2];
    #pragma unroll
    for (int i = 0; i < 2; ++i)
        #pragma unroll
        for (int j = 0; j < 2; ++j) {
            acc[i][j][0] = fzero8(); acc[i][j][1] = fzero8();
        }

    const int NCH = KW * (C2 / 32);          // 5 * 32 = 160 chunks
    // chunk ci -> k-tap = ci>>5, K offset = (ci&31)*32
    auto stage = [&](int ci, int buf) {
        int k  = ci >> 5;
        int kk = (ci & 31) * 32;
        int tl = t0 + srow + k - 2;          // shifted source row
        unsigned short* dstp = &As[buf][srow][sq * 8];
        if (tl >= 0 && tl < TT) {
            const unsigned short* p =
                h1 + ((size_t)(bb << 10) + tl) * C2 + kk + sq * 8;
            async_copy16(dstp, p);
        } else {                             // halo row: zero fill via DS
            uintx4 z = {0u, 0u, 0u, 0u};
            *(uintx4*)dstp = z;
        }
    };

    stage(0, 0);
    wait_async0();
    __syncthreads();

    for (int ci = 0; ci < NCH; ++ci) {
        const int cur = ci & 1;
        if (ci + 1 < NCH) stage(ci + 1, cur ^ 1);

        const int k  = ci >> 5;
        const int kk = (ci & 31) * 32;
        const unsigned short* wk = w2t + (size_t)k * C2 * C2;

        Frag a[2], bl[2], bh[2];
        #pragma unroll
        for (int mi = 0; mi < 2; ++mi) {
            int r = wm * 32 + mi * 16 + lm;
            a[mi].u[0] = *(const uintx4*)(&As[cur][r][lh * 8]);
            a[mi].u[1] = *(const uintx4*)(&As[cur][r][16 + lh * 8]);
        }
        #pragma unroll
        for (int ni = 0; ni < 2; ++ni) {
            int o = n0 + wn * 32 + ni * 16 + lm;
            const unsigned short* wl = wk + (size_t)o * C2 + kk + lh * 8;
            const unsigned short* wh = wk + (size_t)(o + CC) * C2 + kk + lh * 8;
            __builtin_prefetch(wl + 32, 0, 0);
            bl[ni].u[0] = *(const uintx4*)(wl);
            bl[ni].u[1] = *(const uintx4*)(wl + 16);
            bh[ni].u[0] = *(const uintx4*)(wh);
            bh[ni].u[1] = *(const uintx4*)(wh + 16);
        }
        #pragma unroll
        for (int mi = 0; mi < 2; ++mi)
            #pragma unroll
            for (int ni = 0; ni < 2; ++ni) {
                acc[mi][ni][0] = __builtin_amdgcn_wmma_f32_16x16x32_bf16(
                    false, a[mi].v, false, bl[ni].v, (short)0,
                    acc[mi][ni][0], false, false);
                acc[mi][ni][1] = __builtin_amdgcn_wmma_f32_16x16x32_bf16(
                    false, a[mi].v, false, bh[ni].v, (short)0,
                    acc[mi][ni][1], false, false);
            }

        wait_async0();
        __syncthreads();
    }

    #pragma unroll
    for (int mi = 0; mi < 2; ++mi)
        #pragma unroll
        for (int ni = 0; ni < 2; ++ni) {
            int o = n0 + wn * 32 + ni * 16 + lm;
            float biasA = b2[o];
            float biasG = b2[o + CC];
            float scale = bn_g[o] * rsqrtf(bn_var[o] + 1e-5f);
            float shift = bn_b[o] - bn_mean[o] * scale;
            #pragma unroll
            for (int r = 0; r < 8; ++r) {
                int row  = m0 + wm * 32 + mi * 16 + lh * 8 + r;
                float av = acc[mi][ni][0][r] + biasA;
                float gv = acc[mi][ni][1][r] + biasG;
                float v  = av / (1.f + __expf(-gv));         // a * sigmoid(g)
                v = v * scale + shift;                        // BatchNorm (inf)
                h2[(size_t)row * CC + o] = f2bf(v);
            }
        }
}

// ---------------------------------------------------------------------------
// Kernel 4: GEMM3 [BT,512]x[512->512] + bias -> f32 output
// ---------------------------------------------------------------------------
__global__ __launch_bounds__(256) void k_gemm3(
    const unsigned short* __restrict__ h2,   // [BT,512] bf16
    const unsigned short* __restrict__ w3b,  // [512,512] bf16
    const float* __restrict__ b3,
    float* __restrict__ out)                 // [BT,512] f32
{
    __shared__ __align__(16) unsigned short As[2][TILE_ROWS][TILE_STRIDE];

    const int tid  = threadIdx.x;
    const int wave = tid >> 5, lane = tid & 31;
    const int wm = wave >> 2, wn = wave & 3;
    const int lm = lane & 15, lh = lane >> 4;
    const int m0 = blockIdx.x * 64;
    const int n0 = blockIdx.y * 128;
    const int srow = tid >> 2, sq = tid & 3;

    const unsigned short* srcBase = h2 + (size_t)(m0 + srow) * CC + sq * 8;

    floatx8 acc[2][2];
    #pragma unroll
    for (int i = 0; i < 2; ++i)
        #pragma unroll
        for (int j = 0; j < 2; ++j) acc[i][j] = fzero8();

    async_copy16(&As[0][srow][sq * 8], srcBase);
    wait_async0();
    __syncthreads();

    const int NCH = CC / 32;
    for (int c = 0; c < NCH; ++c) {
        const int cur = c & 1;
        if (c + 1 < NCH)
            async_copy16(&As[cur ^ 1][srow][sq * 8], srcBase + (c + 1) * 32);

        Frag a[2], bfr[2];
        #pragma unroll
        for (int mi = 0; mi < 2; ++mi) {
            int r = wm * 32 + mi * 16 + lm;
            a[mi].u[0] = *(const uintx4*)(&As[cur][r][lh * 8]);
            a[mi].u[1] = *(const uintx4*)(&As[cur][r][16 + lh * 8]);
        }
        #pragma unroll
        for (int ni = 0; ni < 2; ++ni) {
            int o = n0 + wn * 32 + ni * 16 + lm;
            const unsigned short* wp = w3b + (size_t)o * CC + c * 32 + lh * 8;
            __builtin_prefetch(wp + 32, 0, 0);
            bfr[ni].u[0] = *(const uintx4*)(wp);
            bfr[ni].u[1] = *(const uintx4*)(wp + 16);
        }
        #pragma unroll
        for (int mi = 0; mi < 2; ++mi)
            #pragma unroll
            for (int ni = 0; ni < 2; ++ni)
                acc[mi][ni] = __builtin_amdgcn_wmma_f32_16x16x32_bf16(
                    false, a[mi].v, false, bfr[ni].v, (short)0, acc[mi][ni],
                    false, false);

        wait_async0();
        __syncthreads();
    }

    #pragma unroll
    for (int mi = 0; mi < 2; ++mi)
        #pragma unroll
        for (int ni = 0; ni < 2; ++ni) {
            int col  = n0 + wn * 32 + ni * 16 + lm;
            float bi = b3[col];
            #pragma unroll
            for (int r = 0; r < 8; ++r) {
                int row = m0 + wm * 32 + mi * 16 + lh * 8 + r;
                out[(size_t)row * CC + col] = acc[mi][ni][r] + bi;
            }
        }
}

// ---------------------------------------------------------------------------
// Host entry
// ---------------------------------------------------------------------------
extern "C" void kernel_launch(void* const* d_in, const int* in_sizes, int n_in,
                              void* d_out, int out_size, void* d_ws, size_t ws_size,
                              hipStream_t stream) {
    (void)in_sizes; (void)n_in; (void)out_size; (void)ws_size;

    const float* x       = (const float*)d_in[0];
    const float* ln_g    = (const float*)d_in[1];
    const float* ln_b    = (const float*)d_in[2];
    const float* w1      = (const float*)d_in[3];
    const float* b1      = (const float*)d_in[4];
    const float* w2      = (const float*)d_in[5];
    const float* b2      = (const float*)d_in[6];
    const float* bn_g    = (const float*)d_in[7];
    const float* bn_b    = (const float*)d_in[8];
    const float* bn_mean = (const float*)d_in[9];
    const float* bn_var  = (const float*)d_in[10];
    const float* w3      = (const float*)d_in[11];
    const float* b3      = (const float*)d_in[12];
    float* out = (float*)d_out;

    char* ws = (char*)d_ws;
    const size_t OFF_W1B = 0;
    const size_t OFF_W3B = OFF_W1B + (size_t)C2 * CC * 2;        //  1.0 MB
    const size_t OFF_W2T = OFF_W3B + (size_t)CC * CC * 2;        //  0.5 MB
    const size_t OFF_XN  = OFF_W2T + (size_t)KW * C2 * C2 * 2;   // 10.0 MB
    const size_t OFF_H1  = OFF_XN  + (size_t)BT * CC * 2;        // 16.0 MB
    const size_t OFF_H2  = OFF_H1  + (size_t)BT * C2 * 2;        // 32.0 MB

    unsigned short* w1b = (unsigned short*)(ws + OFF_W1B);
    unsigned short* w3b = (unsigned short*)(ws + OFF_W3B);
    unsigned short* w2t = (unsigned short*)(ws + OFF_W2T);
    unsigned short* xn  = (unsigned short*)(ws + OFF_XN);
    unsigned short* h1  = (unsigned short*)(ws + OFF_H1);
    unsigned short* h2  = (unsigned short*)(ws + OFF_H2);

    {
        size_t total = (size_t)KW * C2 * C2;                     // 5,242,880
        int blocks = (int)((total + 255) / 256);
        k_prep<<<blocks, 256, 0, stream>>>(w1, w2, w3, w1b, w2t, w3b);
    }
    k_ln<<<BT / 8, 256, 0, stream>>>(x, ln_g, ln_b, xn);
    k_gemm1_silu<<<dim3(BT / 64, C2 / 128), 256, 0, stream>>>(xn, w1b, b1, h1);
    k_conv_glu_bn<<<dim3(BT / 64, CC / 128), 256, 0, stream>>>(
        h1, w2t, b2, bn_g, bn_b, bn_mean, bn_var, h2);
    k_gemm3<<<dim3(BT / 64, CC / 128), 256, 0, stream>>>(h2, w3b, b3, out);
}
